// WeightedMultiHeadAttention_84464826843453
// MI455X (gfx1250) — matrix operand order repaired and checked by reading, not compiled, not run on hardware
//
#include <hip/hip_runtime.h>

// ---------------------------------------------------------------------------
// WeightedMultiHeadAttention for gfx1250 (MI455X), bf16 WMMA pipeline.
// B=2, S=2048, D_MODEL=1024, H=16, D_K=64.  All matmuls -> v_wmma_f32_16x16x32_bf16.
// ---------------------------------------------------------------------------

typedef unsigned int   u32;
typedef unsigned short u16;
typedef __attribute__((ext_vector_type(4)))  u32    u32x4;
typedef __attribute__((ext_vector_type(8)))  float  f32x8;
typedef __attribute__((ext_vector_type(16))) __bf16 bf16x16;

union FragAB {          // 32 bytes: one WMMA A or B operand (16 bf16 per lane)
    bf16x16 bf;
    u32x4   q[2];
};

#define DMODEL 1024
#define SEQ    2048
#define NH     16
#define DK     64
#define BSROWS 4096            // B * S

__device__ __forceinline__ u16 f2bf(float f) {
    union { float f; u32 u; } c; c.f = f;
    u32 r = c.u + 0x7fffu + ((c.u >> 16) & 1u);   // round-to-nearest-even
    return (u16)(r >> 16);
}

// ---------------------------------------------------------------------------
// Elementwise converters
// ---------------------------------------------------------------------------
__global__ void cvt_bf16(const float* __restrict__ in, u16* __restrict__ out, int n) {
    int i = blockIdx.x * blockDim.x + threadIdx.x;
    if (i < n) out[i] = f2bf(in[i]);
}

// W (K x N row-major fp32) -> Wt (N x K row-major bf16)
__global__ void transpose_bf16(const float* __restrict__ in, u16* __restrict__ out) {
    int i = blockIdx.x * blockDim.x + threadIdx.x;      // 1M threads
    int k = i >> 10, n = i & 1023;
    out[n * DMODEL + k] = f2bf(in[i]);
}

__global__ void log_weights(const float* __restrict__ w, float* __restrict__ lw, int n) {
    int i = blockIdx.x * blockDim.x + threadIdx.x;
    if (i < n) lw[i] = __logf(w[i] + 1e-20f);
}

// ---------------------------------------------------------------------------
// Generic bf16 WMMA GEMM:  Y(4096x1024) = X(4096x1024) @ Wt^T + bias
//   X  : bf16 row-major (M x K)
//   Wt : bf16 row-major (N x K)  == W transposed, B-fragments contiguous
// mode 0: out = bf16 (B,H,S,DK) row-major      (Q / K projections)
// mode 1: out = bf16 (B,H,DK,S) (V transposed) (V projection)
// mode 2: out = fp32 row-major (M x N)         (final output proj)
// One wave -> 32(M) x 64(N) tile: B fragments amortized over two M tiles.
// grid (32,16), block 128 (4 waves).
// ---------------------------------------------------------------------------
__global__ void __launch_bounds__(128)
gemm_bf16(const u16* __restrict__ X, const u16* __restrict__ Wt,
          const float* __restrict__ bias, void* __restrict__ out, int mode)
{
    const int lane = threadIdx.x & 31;
    const int wv   = threadIdx.x >> 5;
    const int lr   = lane & 15;       // row (A/C) or column (B) index
    const int lg   = lane >> 4;       // half-wave select
    const int m_base = blockIdx.x * 128 + wv * 32;
    const int n_base = blockIdx.y * 64;

    f32x8 acc[2][4] = {};

    const u16* arow0 = X + (size_t)(m_base + lr) * DMODEL;
    const u16* arow1 = X + (size_t)(m_base + 16 + lr) * DMODEL;

    for (int kb = 0; kb < DMODEL; kb += 32) {
        // ---- load phase: all fragments live simultaneously -> one clause ----
        FragAB a[2], bm[4];
        a[0].q[0] = *(const u32x4*)(arow0 + kb +      lg * 8);
        a[0].q[1] = *(const u32x4*)(arow0 + kb + 16 + lg * 8);
        a[1].q[0] = *(const u32x4*)(arow1 + kb +      lg * 8);
        a[1].q[1] = *(const u32x4*)(arow1 + kb + 16 + lg * 8);
#pragma unroll
        for (int t = 0; t < 4; ++t) {
            const u16* brow = Wt + (size_t)(n_base + t * 16 + lr) * DMODEL + kb + lg * 8;
            bm[t].q[0] = *(const u32x4*)(brow);
            bm[t].q[1] = *(const u32x4*)(brow + 16);
        }
        // ---- compute phase: 8 back-to-back WMMAs ---------------------------
#pragma unroll
        for (int mt = 0; mt < 2; ++mt)
#pragma unroll
            for (int t = 0; t < 4; ++t)
                acc[mt][t] = __builtin_amdgcn_wmma_f32_16x16x32_bf16(
                    false, a[mt].bf, false, bm[t].bf, (short)0, acc[mt][t], false, false);
    }

    // Epilogue: C layout -> VGPR i, lane: row = i + 8*lg, col = lr
#pragma unroll
    for (int mt = 0; mt < 2; ++mt) {
#pragma unroll
        for (int t = 0; t < 4; ++t) {
            const int n  = n_base + t * 16 + lr;
            const float bv = bias[n];
#pragma unroll
            for (int i = 0; i < 8; ++i) {
                const int m = m_base + mt * 16 + i + 8 * lg;
                const float y = acc[mt][t][i] + bv;
                if (mode == 2) {
                    ((float*)out)[(size_t)m * DMODEL + n] = y;
                } else {
                    const int b = m >> 11, s = m & 2047;
                    const int h = n >> 6,  d = n & 63;
                    u16* ob = (u16*)out;
                    if (mode == 0)
                        ob[(((size_t)(b * NH + h) * SEQ + s) << 6) + d] = f2bf(y);  // (B,H,S,DK)
                    else
                        ob[(((size_t)(b * NH + h) * DK + d) << 11) + s] = f2bf(y);  // (B,H,DK,S)
                }
            }
        }
    }
}

// ---------------------------------------------------------------------------
// Flash attention, one wave per 16-row q tile.
//  Qp,Kp: bf16 (B,H,S,DK) row-major.  Vt: bf16 (B,H,DK,S).
//  lw: fp32 log(weights) (B,S).  Out: bf16 (B*S, DMODEL) row-major.
// grid (B*H=32, S/64=32), block 128.
// ---------------------------------------------------------------------------
__global__ void __launch_bounds__(128)
attn_flash(const u16* __restrict__ Qp, const u16* __restrict__ Kp,
           const u16* __restrict__ Vt, const float* __restrict__ lw,
           u16* __restrict__ Out)
{
    __shared__ u16 Pb[4][16 * 32];     // per-wave P tile (C-layout -> A-layout shuffle)

    const int lane = threadIdx.x & 31;
    const int wv   = threadIdx.x >> 5;
    const int lr   = lane & 15;
    const int lg   = lane >> 4;

    const int bh = blockIdx.x;
    const int b  = bh >> 4;
    const int h  = bh & 15;
    const int q_base = blockIdx.y * 64 + wv * 16;

    const u16* Qbh = Qp + (size_t)bh * SEQ * DK;
    const u16* Kbh = Kp + (size_t)bh * SEQ * DK;
    const u16* Vbh = Vt + (size_t)bh * DK * SEQ;
    const float* lwb = lw + b * SEQ;

    // Q tile as two A-fragments (d = 0..31, 32..63); reused for all kv chunks.
    FragAB qa[2];
    {
        const u16* qrow = Qbh + (size_t)(q_base + lr) * DK;
#pragma unroll
        for (int j = 0; j < 2; ++j) {
            qa[j].q[0] = *(const u32x4*)(qrow + j * 32 +      lg * 8);
            qa[j].q[1] = *(const u32x4*)(qrow + j * 32 + 16 + lg * 8);
        }
    }

    float mrow[8], lrow[8];
#pragma unroll
    for (int i = 0; i < 8; ++i) { mrow[i] = -3.0e38f; lrow[i] = 0.0f; }
    f32x8 O[4] = {};

    for (int kb = 0; kb < SEQ; kb += 32) {
        // Stream next K chunk toward the caches (global_prefetch_b8).
        __builtin_prefetch(Kbh + (size_t)(kb + 32 + lr) * DK, 0, 1);

        // ---- load all 4 K fragments, then run the score WMMAs --------------
        FragAB kf[2][2];                       // [kv tile t][d-chunk j]
#pragma unroll
        for (int t = 0; t < 2; ++t) {
            const u16* krow = Kbh + (size_t)(kb + t * 16 + lr) * DK;
#pragma unroll
            for (int j = 0; j < 2; ++j) {
                kf[t][j].q[0] = *(const u32x4*)(krow + j * 32 + lg * 8);
                kf[t][j].q[1] = *(const u32x4*)(krow + j * 32 + lg * 8 + 16);
            }
        }
        f32x8 s[2];
#pragma unroll
        for (int t = 0; t < 2; ++t) {
            f32x8 z = {};
            s[t] = __builtin_amdgcn_wmma_f32_16x16x32_bf16(
                false, qa[0].bf, false, kf[t][0].bf, (short)0, z, false, false);
            s[t] = __builtin_amdgcn_wmma_f32_16x16x32_bf16(
                false, qa[1].bf, false, kf[t][1].bf, (short)0, s[t], false, false);
        }

        const float lw0 = lwb[kb + lr];
        const float lw1 = lwb[kb + 16 + lr];

        // ---- online softmax update (row-wise across 16-lane halves) --------
#pragma unroll
        for (int i = 0; i < 8; ++i) {
            float s0 = s[0][i] * 0.125f + lw0;   // 1/sqrt(64) = 0.125
            float s1 = s[1][i] * 0.125f + lw1;
            float cm = fmaxf(s0, s1);
#pragma unroll
            for (int msk = 1; msk < 16; msk <<= 1)
                cm = fmaxf(cm, __shfl_xor(cm, msk, 32));
            const float mn = fmaxf(mrow[i], cm);
            const float al = __expf(mrow[i] - mn);
            mrow[i] = mn;
            const float p0 = __expf(s0 - mn);
            const float p1 = __expf(s1 - mn);
            float rs = p0 + p1;
#pragma unroll
            for (int msk = 1; msk < 16; msk <<= 1)
                rs += __shfl_xor(rs, msk, 32);
            lrow[i] = lrow[i] * al + rs;
#pragma unroll
            for (int t = 0; t < 4; ++t) O[t][i] *= al;

            const int row = i + 8 * lg;          // C layout -> LDS row-major
            Pb[wv][row * 32 + lr]      = f2bf(p0);
            Pb[wv][row * 32 + 16 + lr] = f2bf(p1);
        }
        __syncthreads();                         // waves are loop-synchronous

        // ---- P (A-layout, from LDS) x Vt (B-layout, contiguous) ------------
        FragAB pf, vf[4];
        {
            const u16* prow = &Pb[wv][lr * 32];
            pf.q[0] = *(const u32x4*)(prow +      lg * 8);
            pf.q[1] = *(const u32x4*)(prow + 16 + lg * 8);
        }
#pragma unroll
        for (int t = 0; t < 4; ++t) {
            const u16* vrow = Vbh + (size_t)(t * 16 + lr) * SEQ + kb + lg * 8;
            vf[t].q[0] = *(const u32x4*)(vrow);
            vf[t].q[1] = *(const u32x4*)(vrow + 16);
        }
#pragma unroll
        for (int t = 0; t < 4; ++t)
            O[t] = __builtin_amdgcn_wmma_f32_16x16x32_bf16(
                false, pf.bf, false, vf[t].bf, (short)0, O[t], false, false);
        __syncthreads();
    }

    // ---- normalize and store (B*S, DMODEL) bf16 ---------------------------
#pragma unroll
    for (int i = 0; i < 8; ++i) {
        const float inv = 1.0f / lrow[i];
        const size_t m = (size_t)(b * SEQ + q_base + i + 8 * lg);
#pragma unroll
        for (int t = 0; t < 4; ++t)
            Out[m * DMODEL + h * DK + t * 16 + lr] = f2bf(O[t][i] * inv);
    }
}

// ---------------------------------------------------------------------------
// Launch
// ---------------------------------------------------------------------------
extern "C" void kernel_launch(void* const* d_in, const int* in_sizes, int n_in,
                              void* d_out, int out_size, void* d_ws, size_t ws_size,
                              hipStream_t stream) {
    const float* query   = (const float*)d_in[0];
    const float* key     = (const float*)d_in[1];
    const float* value   = (const float*)d_in[2];
    const float* weights = (const float*)d_in[3];
    const float* wq = (const float*)d_in[4];  const float* bq = (const float*)d_in[5];
    const float* wk = (const float*)d_in[6];  const float* bk = (const float*)d_in[7];
    const float* wv = (const float*)d_in[8];  const float* bv = (const float*)d_in[9];
    const float* wo = (const float*)d_in[10]; const float* bo = (const float*)d_in[11];

    char* ws = (char*)d_ws;
    const size_t MB = 1u << 20;
    u16*   qb      = (u16*)(ws + 0 * MB);     // 4096x1024 bf16 (8 MB each)
    u16*   kb_     = (u16*)(ws + 8 * MB);
    u16*   vb      = (u16*)(ws + 16 * MB);
    u16*   wqt     = (u16*)(ws + 24 * MB);    // 1024x1024 bf16 transposed (2 MB each)
    u16*   wkt     = (u16*)(ws + 26 * MB);
    u16*   wvt     = (u16*)(ws + 28 * MB);
    u16*   wot     = (u16*)(ws + 30 * MB);
    u16*   Qp      = (u16*)(ws + 32 * MB);    // (B,H,S,DK) bf16
    u16*   Kp      = (u16*)(ws + 40 * MB);    // (B,H,S,DK) bf16
    u16*   Vt      = (u16*)(ws + 48 * MB);    // (B,H,DK,S) bf16
    u16*   AttnOut = (u16*)(ws + 56 * MB);    // (B*S, DMODEL) bf16
    float* logw    = (float*)(ws + 64 * MB);  // (B,S) fp32

    const int nAct = BSROWS * DMODEL;         // 4,194,304
    const int nW   = DMODEL * DMODEL;         // 1,048,576

    cvt_bf16<<<nAct / 256, 256, 0, stream>>>(query, qb, nAct);
    cvt_bf16<<<nAct / 256, 256, 0, stream>>>(key,   kb_, nAct);
    cvt_bf16<<<nAct / 256, 256, 0, stream>>>(value, vb, nAct);
    transpose_bf16<<<nW / 256, 256, 0, stream>>>(wq, wqt);
    transpose_bf16<<<nW / 256, 256, 0, stream>>>(wk, wkt);
    transpose_bf16<<<nW / 256, 256, 0, stream>>>(wv, wvt);
    transpose_bf16<<<nW / 256, 256, 0, stream>>>(wo, wot);
    log_weights<<<(2 * SEQ) / 256, 256, 0, stream>>>(weights, logw, 2 * SEQ);

    dim3 gg(32, 16), gb(128);
    gemm_bf16<<<gg, gb, 0, stream>>>(qb,  wqt, bq, (void*)Qp, 0);
    gemm_bf16<<<gg, gb, 0, stream>>>(kb_, wkt, bk, (void*)Kp, 0);
    gemm_bf16<<<gg, gb, 0, stream>>>(vb,  wvt, bv, (void*)Vt, 1);

    attn_flash<<<dim3(32, 32), 128, 0, stream>>>(Qp, Kp, Vt, logw, AttnOut);

    gemm_bf16<<<gg, gb, 0, stream>>>(AttnOut, wot, bo, d_out, 2);
}
